// MViTBlock_28578712388189
// MI455X (gfx1250) — matrix operand order repaired and checked by reading, not compiled, not run on hardware
//
#include <hip/hip_runtime.h>
#include <hip/hip_bf16.h>

typedef __bf16 bf16;
typedef __attribute__((ext_vector_type(16))) __bf16 v16bf;
typedef __attribute__((ext_vector_type(8)))  float  v8f;
typedef __attribute__((ext_vector_type(4)))  uint32_t u32x4;
typedef __attribute__((ext_vector_type(8)))  uint32_t u32x8;

#define B_   8
#define N_   3136
#define C_   768
#define H_   12
#define HD_  64
#define NP_  1046
#define NPAD_ 1088     // 17 * 64
#define HID_ 3072
#define M_   (B_ * N_) // 25088

// ---------------- CDNA5 async / WMMA helpers ----------------

// async global -> LDS, 16B per lane, tracked by ASYNCcnt
__device__ __forceinline__ void async_copy_b128(void* lds, const void* g) {
    asm volatile("global_load_async_to_lds_b128 %0, %1, off"
                 :: "v"((uint32_t)(uintptr_t)lds), "v"(g)
                 : "memory");
}
#define WAIT_ASYNC(n) asm volatile("s_wait_asynccnt %0" :: "i"(n) : "memory")

__device__ __forceinline__ v8f wmma_bf16(v16bf a, v16bf b, v8f c) {
    return __builtin_amdgcn_wmma_f32_16x16x32_bf16(false, a, false, b, (short)0, c,
                                                   false, false);
}

// A-matrix 16x32 bf16 tile from row-major memory (row stride ld elements).
// lane (0..15): M=lane, elem e -> K=16*hi+e
__device__ __forceinline__ v16bf load_tile_a(const bf16* base, int ld) {
    const int lane = threadIdx.x & 31;
    const int r = lane & 15;
    const int hi = lane >> 4;
    const bf16* p = base + r * ld + 16 * hi;
    v16bf t;
#pragma unroll
    for (int j = 0; j < 16; ++j) t[j] = p[j];
    return t;
}

// B-matrix 32x16 bf16 tile, supplied as 16 rows (N) x 32 (K) row-major.
// lane&15 = N, elem e -> K = 16*hi + e
__device__ __forceinline__ v16bf load_tile_b(const bf16* base, int ld) {
    const int lane = threadIdx.x & 31;
    const int n = lane & 15;
    const int hi = lane >> 4;
    const bf16* p = base + n * ld + 16 * hi;
    v16bf t;
#pragma unroll
    for (int j = 0; j < 16; ++j) t[j] = p[j];
    return t;
}

// ---------------- small utility kernels ----------------

__global__ __launch_bounds__(256) void cvt_f32_bf16(const float* __restrict__ in,
                                                    bf16* __restrict__ out, int n) {
    int i = blockIdx.x * 256 + threadIdx.x;
    if (i < n) out[i] = (bf16)in[i];
}

// LayerNorm over last dim (768), one block per row, output bf16
__global__ __launch_bounds__(256) void ln_kernel(const float* __restrict__ x,
                                                 const float* __restrict__ g,
                                                 const float* __restrict__ b,
                                                 bf16* __restrict__ out) {
    __shared__ float r1[256], r2[256];
    const int row = blockIdx.x;
    const int t = threadIdx.x;
    const float* xr = x + (size_t)row * C_;
    float v[3], s = 0.f, ss = 0.f;
#pragma unroll
    for (int i = 0; i < 3; ++i) {
        v[i] = xr[t + 256 * i];
        s += v[i];
        ss += v[i] * v[i];
    }
    r1[t] = s; r2[t] = ss;
    __syncthreads();
    for (int o = 128; o > 0; o >>= 1) {
        if (t < o) { r1[t] += r1[t + o]; r2[t] += r2[t + o]; }
        __syncthreads();
    }
    const float mean = r1[0] * (1.f / C_);
    const float var  = r2[0] * (1.f / C_) - mean * mean;
    const float inv  = rsqrtf(var + 1e-5f);
#pragma unroll
    for (int i = 0; i < 3; ++i) {
        int c = t + 256 * i;
        out[(size_t)row * C_ + c] = (bf16)((v[i] - mean) * inv * g[c] + b[c]);
    }
}

// AvgPool1d(kernel=3, stride=3, pad=1, count_include_pad) over seq for k and v,
// writing zero-padded (NPAD_) per-(b,h) bf16 buffers.
__global__ __launch_bounds__(256) void pool_kernel(const bf16* __restrict__ qkv,
                                                   bf16* __restrict__ kp,
                                                   bf16* __restrict__ vp) {
    const int idx = blockIdx.x * 256 + threadIdx.x;
    if (idx >= B_ * H_ * NPAD_ * HD_) return;
    const int d = idx & (HD_ - 1);
    const int p = (idx >> 6) % NPAD_;
    const int bh = idx / (NPAD_ * HD_);
    const int h = bh % H_, b = bh / H_;
    float sk = 0.f, sv = 0.f;
    if (p < NP_) {
#pragma unroll
        for (int u = 0; u < 3; ++u) {
            int n = 3 * p - 1 + u;
            if (n >= 0 && n < N_) {
                const bf16* rowp = qkv + (size_t)(b * N_ + n) * (3 * C_) + h * HD_ + d;
                sk += (float)rowp[C_];
                sv += (float)rowp[2 * C_];
            }
        }
        sk *= (1.f / 3.f); sv *= (1.f / 3.f);
    }
    kp[idx] = (bf16)sk;
    vp[idx] = (bf16)sv;
}

// ---------------- generic bf16 WMMA GEMM: out = A(MxK) * W(NoutxK)^T ----------------
// Double-buffered LDS, async global->LDS copies overlap the WMMAs.
// MODE 0: outh = bf16(acc)
// MODE 1: outf = resid + acc + bias
// MODE 2: outh = bf16(gelu(acc + bias))
// MODE 3: outf += acc + bias
template <int MODE>
__global__ __launch_bounds__(256) void gemm_kernel(const bf16* __restrict__ A,
                                                   const bf16* __restrict__ W,
                                                   const float* __restrict__ bias,
                                                   const float* __restrict__ resid,
                                                   float* __restrict__ outf,
                                                   bf16* __restrict__ outh,
                                                   int M, int Nout, int K) {
    __shared__ bf16 sA[2][128 * 32];
    __shared__ bf16 sW[2][128 * 32];
    const int t = threadIdx.x;
    const int wave = t >> 5;
    const int lane = t & 31;
    const int mbase = blockIdx.y * 128;
    const int nbase = blockIdx.x * 128;
    const int wm = (wave >> 1) * 32;  // 0,32,64,96
    const int wn = (wave & 1) * 64;   // 0,64

    v8f acc[2][4];
#pragma unroll
    for (int i = 0; i < 2; ++i)
#pragma unroll
        for (int j = 0; j < 4; ++j)
#pragma unroll
            for (int e = 0; e < 8; ++e) acc[i][j][e] = 0.f;

    const int lr = t >> 1;          // 0..127: tile row this thread copies
    const int lc = (t & 1) * 16;    // 16-element chunk

    // 4 async b128 per thread per k-step (2 for A, 2 for W) -> ASYNCcnt += 4 per wave
    auto issue_tile = [&](int kt, int buf) {
        const bf16* ga = A + (size_t)(mbase + lr) * K + kt * 32 + lc;
        async_copy_b128(&sA[buf][lr * 32 + lc], ga);
        async_copy_b128(&sA[buf][lr * 32 + lc + 8], ga + 8);
        const bf16* gw = W + (size_t)(nbase + lr) * K + kt * 32 + lc;
        async_copy_b128(&sW[buf][lr * 32 + lc], gw);
        async_copy_b128(&sW[buf][lr * 32 + lc + 8], gw + 8);
    };

    const int ktiles = K / 32;
    issue_tile(0, 0);
    for (int kt = 0; kt < ktiles; ++kt) {
        const int cur = kt & 1;
        if (kt + 1 < ktiles) {
            issue_tile(kt + 1, cur ^ 1);   // prefetch next tile into other buffer
            WAIT_ASYNC(4);                 // in-order: drains current tile's 4 copies
        } else {
            WAIT_ASYNC(0);
        }
        __syncthreads();
        v16bf a0 = load_tile_a(sA[cur] + (wm + 0) * 32, 32);
        v16bf a1 = load_tile_a(sA[cur] + (wm + 16) * 32, 32);
#pragma unroll
        for (int j = 0; j < 4; ++j) {
            v16bf bb = load_tile_b(sW[cur] + (wn + 16 * j) * 32, 32);
            acc[0][j] = wmma_bf16(a0, bb, acc[0][j]);
            acc[1][j] = wmma_bf16(a1, bb, acc[1][j]);
        }
        __syncthreads();
    }

    const int cn = lane & 15;
    const int hi = lane >> 4;
#pragma unroll
    for (int i = 0; i < 2; ++i) {
#pragma unroll
        for (int j = 0; j < 4; ++j) {
            const int gn = nbase + wn + 16 * j + cn;
#pragma unroll
            for (int e = 0; e < 8; ++e) {
                const int gm = mbase + wm + 16 * i + e + 8 * hi;
                const size_t o = (size_t)gm * Nout + gn;
                float v = acc[i][j][e];
                if constexpr (MODE == 0) {
                    outh[o] = (bf16)v;
                } else if constexpr (MODE == 1) {
                    outf[o] = resid[o] + v + bias[gn];
                } else if constexpr (MODE == 2) {
                    v += bias[gn];
                    outh[o] = (bf16)(0.5f * v * (1.0f + erff(v * 0.70710678f)));
                } else {
                    outf[o] += v + bias[gn];
                }
            }
        }
    }
}

// ---------------- attention: block = (b, h, 64 q-rows); scores in LDS ----------------
__global__ __launch_bounds__(256) void attn_kernel(const bf16* __restrict__ qkv,
                                                   const bf16* __restrict__ kp,
                                                   const bf16* __restrict__ vp,
                                                   bf16* __restrict__ o) {
    __shared__ bf16 sS[64 * NPAD_];   // 136 KB scores (bf16, exp'd in place)
    __shared__ bf16 sQ[64 * 64];
    __shared__ bf16 sKV[64 * 64];
    __shared__ float sRed[256];
    __shared__ float sMax[64];
    __shared__ float sSum[64];

    const int t = threadIdx.x;
    const int wave = t >> 5;
    const int lane = t & 31;
    const int qbase = blockIdx.x * 64;
    const int h = blockIdx.y;
    const int b = blockIdx.z;
    const size_t bh = (size_t)(b * H_ + h);

    // ---- Q tile (64 rows x 128B, row stride 4608B) via Tensor Data Mover ----
    if (wave == 0) {
        const uint64_t gaddr =
            (uint64_t)(uintptr_t)(qkv + (size_t)(b * N_ + qbase) * (3 * C_) + h * HD_);
        u32x4 g0;
        g0[0] = 1u;                                   // count=1, user descriptor
        g0[1] = (uint32_t)(uintptr_t)&sQ[0];          // lds_addr (bytes)
        g0[2] = (uint32_t)gaddr;                      // global_addr[31:0]
        g0[3] = (uint32_t)((gaddr >> 32) & 0x1FFFFFFu) | (2u << 30);  // addr[56:32], type=2
        const uint64_t TD0 = (3 * C_ * 2) / 8;        // tensor dim0 = 576 (8B units)
        const uint64_t TD1 = 1u << 20;                // plenty of rows (no OOB needed)
        const uint64_t TILE0 = (HD_ * 2) / 8;         // 16 (8B units per tile row)
        const uint64_t TILE1 = 64, TILE2 = 0;
        const uint64_t S0 = TD0, S1 = 0;              // dim0 stride; dim1 stride unused
        const uint64_t q0 = (3ull << 16) | ((TD0 & 0xFFFFull) << 48);   // data_size=8B
        const uint64_t q1 = (TD0 >> 16) | (TD1 << 16) | (TILE0 << 48);
        const uint64_t q2 = TILE1 | (TILE2 << 16) | ((S0 & 0xFFFFFFFFull) << 32);
        const uint64_t q3 = (S0 >> 32) | (S1 << 16);
        u32x8 g1;
        g1[0] = (uint32_t)q0; g1[1] = (uint32_t)(q0 >> 32);
        g1[2] = (uint32_t)q1; g1[3] = (uint32_t)(q1 >> 32);
        g1[4] = (uint32_t)q2; g1[5] = (uint32_t)(q2 >> 32);
        g1[6] = (uint32_t)q3; g1[7] = (uint32_t)(q3 >> 32);
        asm volatile("tensor_load_to_lds %0, %1" :: "s"(g0), "s"(g1) : "memory");
        __builtin_amdgcn_s_wait_tensorcnt(0);
    }

    const int qsub = (wave >> 1) * 16;  // 0..48
    const int nh = (wave & 1) * 32;     // 0,32
    const int cn = lane & 15;
    const int hi = lane >> 4;

    // -------- phase 1: S = (Q K^T) * 1/8, masked, stored bf16 in LDS --------
    for (int kt = 0; kt < NPAD_ / 64; ++kt) {
        const int kvbase = kt * 64;
        __syncthreads();   // prior WMMA reads of sKV done; Q publish on kt==0
        {
            const int r = t >> 2, dbase = (t & 3) * 16;
            const bf16* g = kp + bh * (NPAD_ * HD_) + (size_t)(kvbase + r) * HD_ + dbase;
            async_copy_b128(&sKV[r * 64 + dbase], g);
            async_copy_b128(&sKV[r * 64 + dbase + 8], g + 8);
        }
        WAIT_ASYNC(0);
        __syncthreads();
        v8f acc[2];
#pragma unroll
        for (int j = 0; j < 2; ++j)
#pragma unroll
            for (int e = 0; e < 8; ++e) acc[j][e] = 0.f;
#pragma unroll
        for (int ks = 0; ks < 2; ++ks) {
            v16bf a = load_tile_a(sQ + qsub * 64 + ks * 32, 64);
#pragma unroll
            for (int j = 0; j < 2; ++j) {
                v16bf bb = load_tile_b(sKV + (nh + 16 * j) * 64 + ks * 32, 64);
                acc[j] = wmma_bf16(a, bb, acc[j]);
            }
        }
#pragma unroll
        for (int j = 0; j < 2; ++j) {
            const int col = kvbase + nh + 16 * j + cn;
            const bool valid = col < NP_;
#pragma unroll
            for (int e = 0; e < 8; ++e) {
                float s = acc[j][e] * 0.125f;  // hd^-0.5 = 1/8
                if (!valid) s = -3.0e38f;
                sS[(qsub + e + 8 * hi) * NPAD_ + col] = (bf16)s;
            }
        }
    }
    __syncthreads();

    // -------- phase 2: softmax (two-pass over 1088 cols, 4 threads/row) --------
    {
        const int row = t >> 2, seg = t & 3;
        bf16* srow = sS + row * NPAD_ + seg * (NPAD_ / 4);
        float mx = -3.0e38f;
        for (int i = 0; i < NPAD_ / 4; ++i) mx = fmaxf(mx, (float)srow[i]);
        sRed[t] = mx;
        __syncthreads();
        if (seg == 0)
            sMax[row] = fmaxf(fmaxf(sRed[t], sRed[t + 1]), fmaxf(sRed[t + 2], sRed[t + 3]));
        __syncthreads();
        const float m = sMax[row];
        float sum = 0.f;
        for (int i = 0; i < NPAD_ / 4; ++i) {
            float p = __expf((float)srow[i] - m);
            sum += p;
            srow[i] = (bf16)p;
        }
        sRed[t] = sum;
        __syncthreads();
        if (seg == 0) sSum[row] = sRed[t] + sRed[t + 1] + sRed[t + 2] + sRed[t + 3];
        __syncthreads();
    }

    // -------- phase 3: O = P V (V transposed into LDS), normalize, store --------
    v8f oacc[2];
#pragma unroll
    for (int j = 0; j < 2; ++j)
#pragma unroll
        for (int e = 0; e < 8; ++e) oacc[j][e] = 0.f;

    for (int kt = 0; kt < NPAD_ / 64; ++kt) {
        const int kvbase = kt * 64;
        __syncthreads();
        {
            const int r = t & 63, dbase = (t >> 6) * 16;
            const bf16* g = vp + bh * (NPAD_ * HD_) + (size_t)(kvbase + r) * HD_ + dbase;
#pragma unroll
            for (int i = 0; i < 16; ++i) sKV[(dbase + i) * 64 + r] = g[i];
        }
        __syncthreads();
#pragma unroll
        for (int ks = 0; ks < 2; ++ks) {
            v16bf a = load_tile_a(sS + qsub * NPAD_ + kvbase + ks * 32, NPAD_);
#pragma unroll
            for (int j = 0; j < 2; ++j) {
                v16bf bb = load_tile_b(sKV + (nh + 16 * j) * 64 + ks * 32, 64);
                oacc[j] = wmma_bf16(a, bb, oacc[j]);
            }
        }
    }
#pragma unroll
    for (int j = 0; j < 2; ++j) {
        const int d = nh + 16 * j + cn;
#pragma unroll
        for (int e = 0; e < 8; ++e) {
            const int r = qsub + e + 8 * hi;
            const float val = oacc[j][e] / sSum[r];
            o[(size_t)(b * N_ + qbase + r) * C_ + h * HD_ + d] = (bf16)val;
        }
    }
}

// ---------------- launcher ----------------

extern "C" void kernel_launch(void* const* d_in, const int* in_sizes, int n_in,
                              void* d_out, int out_size, void* d_ws, size_t ws_size,
                              hipStream_t stream) {
    const float* x      = (const float*)d_in[0];
    const float* ln1_g  = (const float*)d_in[1];
    const float* ln1_b  = (const float*)d_in[2];
    const float* qkv_w  = (const float*)d_in[3];
    const float* proj_w = (const float*)d_in[4];
    const float* proj_b = (const float*)d_in[5];
    const float* ln2_g  = (const float*)d_in[6];
    const float* ln2_b  = (const float*)d_in[7];
    const float* fc1_w  = (const float*)d_in[8];
    const float* fc1_b  = (const float*)d_in[9];
    const float* fc2_w  = (const float*)d_in[10];
    const float* fc2_b  = (const float*)d_in[11];
    float* out = (float*)d_out;

    char* ws = (char*)d_ws;
    size_t off = 0;
    auto alloc = [&](size_t bytes) {
        void* p = ws + off;
        off += (bytes + 255) & ~(size_t)255;
        return p;
    };
    bf16* wqkv = (bf16*)alloc((size_t)3 * C_ * C_ * 2);
    bf16* wproj = (bf16*)alloc((size_t)C_ * C_ * 2);
    bf16* wfc1 = (bf16*)alloc((size_t)HID_ * C_ * 2);
    bf16* wfc2 = (bf16*)alloc((size_t)C_ * HID_ * 2);
    bf16* act0 = (bf16*)alloc((size_t)M_ * C_ * 2);          // ln1 / attn-out / ln2
    bf16* big  = (bf16*)alloc((size_t)M_ * HID_ * 2);        // qkv, then mlp hidden
    bf16* kp   = (bf16*)alloc((size_t)B_ * H_ * NPAD_ * HD_ * 2);
    bf16* vp   = (bf16*)alloc((size_t)B_ * H_ * NPAD_ * HD_ * 2);

    // weights f32 -> bf16
    {
        int n;
        n = 3 * C_ * C_; cvt_f32_bf16<<<(n + 255) / 256, 256, 0, stream>>>(qkv_w, wqkv, n);
        n = C_ * C_;     cvt_f32_bf16<<<(n + 255) / 256, 256, 0, stream>>>(proj_w, wproj, n);
        n = HID_ * C_;   cvt_f32_bf16<<<(n + 255) / 256, 256, 0, stream>>>(fc1_w, wfc1, n);
        n = C_ * HID_;   cvt_f32_bf16<<<(n + 255) / 256, 256, 0, stream>>>(fc2_w, wfc2, n);
    }

    // LN1
    ln_kernel<<<M_, 256, 0, stream>>>(x, ln1_g, ln1_b, act0);

    // QKV: (M x 768) x (2304 x 768)^T -> bf16
    gemm_kernel<0><<<dim3(3 * C_ / 128, M_ / 128), 256, 0, stream>>>(
        act0, wqkv, nullptr, nullptr, nullptr, big, M_, 3 * C_, C_);

    // avgpool K/V into padded buffers
    {
        int n = B_ * H_ * NPAD_ * HD_;
        pool_kernel<<<(n + 255) / 256, 256, 0, stream>>>(big, kp, vp);
    }

    // attention -> act0 (bf16, (M x 768))
    attn_kernel<<<dim3(N_ / 64, H_, B_), 256, 0, stream>>>(big, kp, vp, act0);

    // proj + bias + residual -> d_out (f32)
    gemm_kernel<1><<<dim3(C_ / 128, M_ / 128), 256, 0, stream>>>(
        act0, wproj, proj_b, x, out, nullptr, M_, C_, C_);

    // LN2 -> act0
    ln_kernel<<<M_, 256, 0, stream>>>(out, ln2_g, ln2_b, act0);

    // fc1 + bias + GELU -> big (bf16, M x 3072)
    gemm_kernel<2><<<dim3(HID_ / 128, M_ / 128), 256, 0, stream>>>(
        act0, wfc1, fc1_b, nullptr, nullptr, big, M_, HID_, C_);

    // fc2 + bias, accumulate into d_out
    gemm_kernel<3><<<dim3(C_ / 128, M_ / 128), 256, 0, stream>>>(
        big, wfc2, fc2_b, nullptr, out, nullptr, M_, C_, HID_);
}
